// PairwiseFeatureBatch_69930657513704
// MI455X (gfx1250) — matrix operand order repaired and checked by reading, not compiled, not run on hardware
//
#include <hip/hip_runtime.h>
#include <math.h>

// ---------------------------------------------------------------------------
// Types / WMMA helpers (CDNA5 gfx1250, wave32)
// ---------------------------------------------------------------------------
typedef __attribute__((ext_vector_type(16))) _Float16 v16h;
typedef __attribute__((ext_vector_type(8)))  _Float16 v8h;
typedef __attribute__((ext_vector_type(8)))  float    v8f;

union V16U { v16h v; v8h h[2]; };

// A-operand fragment (16x32 f16). Per ISA: lanes 0-15 hold row M=lane,
// K = {ks+0..7, ks+16..23}; lanes 16-31 hold row M=lane-16,
// K = {ks+8..15, ks+24..31}.  row points at a K-contiguous row in LDS.
__device__ __forceinline__ v16h ldA16(const _Float16* row, int ks, int kh) {
  V16U u;
  u.h[0] = *(const v8h*)(row + ks + kh * 8);
  u.h[1] = *(const v8h*)(row + ks + 16 + kh * 8);
  return u.v;
}
// B-operand fragment (32x16 f16). Lane holds column (lane&15), K order same
// as A: K = {ks+8*kh+0..7, ks+16+8*kh+0..7}. col points at column-major weights
// (i.e. row n of W[out][K]).
__device__ __forceinline__ v16h ldB16(const _Float16* col, int ks, int kh) {
  V16U u;
  u.h[0] = *(const v8h*)(col + ks + kh * 8);
  u.h[1] = *(const v8h*)(col + ks + 16 + kh * 8);
  return u.v;
}
__device__ __forceinline__ v8f wmma16(v16h a, v16h b, v8f c) {
  return __builtin_amdgcn_wmma_f32_16x16x32_f16(false, a, false, b,
                                                (short)0, c, false, false);
}
__device__ __forceinline__ v8f vzero8() {
  v8f z; for (int i = 0; i < 8; ++i) z[i] = 0.0f; return z;
}
__device__ __forceinline__ float gelu_exact(float v) {
  return 0.5f * v * (1.0f + erff(v * 0.7071067811865475f));
}

// Problem constants
#define NN 512
#define BB 2
#define DD 128
#define OO 16
#define HH 64
#define CH 48
#define NPAIR 131328            // 512*513/2
#define MROWS (NPAIR * BB)      // 262656, /16 = 16416 tiles

// ---------------------------------------------------------------------------
// prep: weight conversions f32 -> f16 (+ conv weight repack, K-padded)
// ---------------------------------------------------------------------------
__global__ void prep_weights_kernel(
    const float* __restrict__ w1, const float* __restrict__ w2,
    const float* __restrict__ w3, const float* __restrict__ sw1,
    const float* __restrict__ sw2, const float* __restrict__ sw3,
    const float* __restrict__ cw1, const float* __restrict__ cw2,
    _Float16* __restrict__ w1h, _Float16* __restrict__ w2h,
    _Float16* __restrict__ w3h, _Float16* __restrict__ sw1h,
    _Float16* __restrict__ sw2h, _Float16* __restrict__ sw3h,
    _Float16* __restrict__ wB1, _Float16* __restrict__ wB2) {
  int i = blockIdx.x * 256 + threadIdx.x;
  if (i < HH * 6 * DD) w1h[i] = (_Float16)w1[i];
  if (i < HH * HH)     w2h[i] = (_Float16)w2[i];
  if (i < OO * HH)     w3h[i] = (_Float16)w3[i];
  if (i < HH * 3 * DD) sw1h[i] = (_Float16)sw1[i];
  if (i < HH * HH)     sw2h[i] = (_Float16)sw2[i];
  if (i < OO * HH)     sw3h[i] = (_Float16)sw3[i];
  if (i < CH * 160) {             // conv1 weights: [co][k], k=(ky*3+kx)*16+ci
    int co = i / 160, k = i % 160;
    _Float16 v = (_Float16)0.0f;
    if (k < 144) { int kidx = k / 16, ci = k % 16;
      v = (_Float16)cw1[co * 144 + ci * 9 + kidx]; }
    wB1[i] = v;
  }
  if (i < OO * 448) {             // conv2 weights: [co][k], k=(ky*3+kx)*48+ci
    int co = i / 448, k = i % 448;
    _Float16 v = (_Float16)0.0f;
    if (k < 432) { int kidx = k / 48, ci = k % 48;
      v = (_Float16)cw2[co * 432 + ci * 9 + kidx]; }
    wB2[i] = v;
  }
}

// ---------------------------------------------------------------------------
// prefix sums of x, x^2, x^3 along N, with leading zero row. [513][B][D]
// ---------------------------------------------------------------------------
__global__ void prefix_kernel(const float* __restrict__ x,
                              float* __restrict__ xc,
                              float* __restrict__ x2c,
                              float* __restrict__ x3c) {
  int col = blockIdx.x * blockDim.x + threadIdx.x;  // 0..255 == (b,d)
  if (col >= BB * DD) return;
  float s1 = 0.f, s2 = 0.f, s3 = 0.f;
  xc[col] = 0.f; x2c[col] = 0.f; x3c[col] = 0.f;
  for (int i = 0; i < NN; ++i) {
    float v = x[i * (BB * DD) + col];
    s1 += v; s2 += v * v; s3 += v * v * v;
    xc[(i + 1) * (BB * DD) + col] = s1;
    x2c[(i + 1) * (BB * DD) + col] = s2;
    x3c[(i + 1) * (BB * DD) + col] = s3;
  }
}

// ---------------------------------------------------------------------------
// zero f16 buffer (dense S, upper triangle must stay zero)
// ---------------------------------------------------------------------------
__global__ void zero32_kernel(unsigned int* __restrict__ p, unsigned int n) {
  unsigned int i = blockIdx.x * blockDim.x + threadIdx.x;
  unsigned int stride = gridDim.x * blockDim.x;
  for (; i < n; i += stride) p[i] = 0u;
}

// ---------------------------------------------------------------------------
// Pair MLP: build 6*128 features on the fly per 16-row tile, 768->64->64->16
// with f16 WMMA / f32 accum; scatter into dense S16 lower triangle.
// ---------------------------------------------------------------------------
#define PM_WAVES 8
__global__ __launch_bounds__(256) void pair_mlp_kernel(
    const float* __restrict__ x, const float* __restrict__ xc,
    const float* __restrict__ x2c, const float* __restrict__ x3c,
    const _Float16* __restrict__ w1h, const float* __restrict__ b1,
    const _Float16* __restrict__ w2h, const float* __restrict__ b2,
    const _Float16* __restrict__ w3h, const float* __restrict__ b3,
    _Float16* __restrict__ S16) {
  __shared__ _Float16 sA[PM_WAVES][16 * 128];
  __shared__ _Float16 sH[PM_WAVES][16 * 64];
  __shared__ int sRi[PM_WAVES][16];
  __shared__ int sCi[PM_WAVES][16];

  const int wave = threadIdx.x >> 5;
  const int lane = threadIdx.x & 31;
  const int kh = lane >> 4;
  const int ln = lane & 15;
  const int tile = blockIdx.x * PM_WAVES + wave;  // exact: 2052*8 = 16416

  _Float16* A = sA[wave];
  _Float16* Hb = sH[wave];

  // per-tile pair metadata (tril row-major ordering: row r, col 0..r)
  if (lane < 16) {
    int gidx = tile * 16 + lane;
    int p = gidx >> 1;  // flat row = p*B + b
    float t = sqrtf(8.0f * (float)p + 1.0f);
    int r = (int)((t - 1.0f) * 0.5f);
    while (((r + 1) * (r + 2)) / 2 <= p) ++r;
    while ((r * (r + 1)) / 2 > p) --r;
    sRi[wave][lane] = r;
    sCi[wave][lane] = p - (r * (r + 1)) / 2;
  }

  v8f acc[4]; for (int t = 0; t < 4; ++t) acc[t] = vzero8();

  // Layer 1: K = 6 segments x 128
  for (int seg = 0; seg < 6; ++seg) {
    for (int i = 0; i < 64; ++i) {  // 16x128 elements, 32 lanes
      int idx = i * 32 + lane;
      int m = idx >> 7, c = idx & 127;
      int ri = sRi[wave][m], ci = sCi[wave][m];
      int b = m & 1;
      float v;
      if (seg < 3) {
        float xa = x[(ri * BB + b) * DD + c];
        float xb = x[(ci * BB + b) * DD + c];
        v = (seg == 0) ? xa : ((seg == 1) ? xb : xa * xb);
      } else {
        const float* cums = (seg == 3) ? xc : ((seg == 4) ? x2c : x3c);
        float L = (float)(ri - ci + 1);
        v = (cums[((ri + 1) * BB + b) * DD + c] -
             cums[(ci * BB + b) * DD + c]) / L;
      }
      A[m * 128 + c] = (_Float16)v;
    }
    for (int kk = 0; kk < 4; ++kk) {
      v16h a = ldA16(A + ln * 128, kk * 32, kh);
      for (int t = 0; t < 4; ++t) {
        v16h bm = ldB16(w1h + (t * 16 + ln) * 768, seg * 128 + kk * 32, kh);
        acc[t] = wmma16(a, bm, acc[t]);
      }
    }
  }
  // bias + GELU -> Hb
  for (int t = 0; t < 4; ++t) {
    int n = t * 16 + ln;
    float bias = b1[n];
    for (int j = 0; j < 8; ++j) {
      int m = j + kh * 8;
      Hb[m * 64 + n] = (_Float16)gelu_exact(acc[t][j] + bias);
    }
  }
  // Layer 2: 64 -> 64
  v8f acc2[4]; for (int t = 0; t < 4; ++t) acc2[t] = vzero8();
  for (int kk = 0; kk < 2; ++kk) {
    v16h a = ldA16(Hb + ln * 64, kk * 32, kh);
    for (int t = 0; t < 4; ++t) {
      v16h bm = ldB16(w2h + (t * 16 + ln) * 64, kk * 32, kh);
      acc2[t] = wmma16(a, bm, acc2[t]);
    }
  }
  for (int t = 0; t < 4; ++t) {
    int n = t * 16 + ln;
    float bias = b2[n];
    for (int j = 0; j < 8; ++j) {
      int m = j + kh * 8;
      Hb[m * 64 + n] = (_Float16)gelu_exact(acc2[t][j] + bias);
    }
  }
  // Layer 3: 64 -> 16
  v8f acc3 = vzero8();
  for (int kk = 0; kk < 2; ++kk) {
    v16h a = ldA16(Hb + ln * 64, kk * 32, kh);
    v16h bm = ldB16(w3h + ln * 64, kk * 32, kh);
    acc3 = wmma16(a, bm, acc3);
  }
  float bias3 = b3[ln];
  for (int j = 0; j < 8; ++j) {
    int m = j + kh * 8;
    int ri = sRi[wave][m], ci = sCi[wave][m], b = m & 1;
    S16[(((size_t)ri * NN + ci) * BB + b) * OO + ln] =
        (_Float16)(acc3[j] + bias3);
  }
}

// ---------------------------------------------------------------------------
// Conv1 as implicit GEMM: 16 output pixels/wave, K=9*16 (pad 160), N=48.
// Input S16 [512][512][B][16] (channels contiguous), pad=2, out 514x514,
// GELU fused, output H16 [514][514][B][48] f16.
// ---------------------------------------------------------------------------
#define C1_WAVES 4
__global__ __launch_bounds__(128) void conv1_kernel(
    const _Float16* __restrict__ S16, const _Float16* __restrict__ wB1,
    const float* __restrict__ cb1, _Float16* __restrict__ H16) {
  __shared__ _Float16 sA[C1_WAVES][16 * 160];
  const int wave = threadIdx.x >> 5;
  const int lane = threadIdx.x & 31;
  const int kh = lane >> 4, ln = lane & 15;
  int wid = blockIdx.x * C1_WAVES + wave;  // exact: 8481*4 = 514*33*2
  int b = wid & 1;
  int rest = wid >> 1;
  int xt = rest % 33;
  int yo = rest / 33;
  int xbase = xt * 16;
  _Float16* A = sA[wave];

  unsigned int* Az = (unsigned int*)A;    // zero whole tile (padding + K tail)
  for (int i = 0; i < 40; ++i) Az[i * 32 + lane] = 0u;

  for (int i = 0; i < 5; ++i) {           // 144 im2col chunk jobs
    int j = i * 32 + lane;
    if (j < 144) {
      int m = j / 9, kidx = j % 9;
      int ky = kidx / 3, kx = kidx % 3;
      int yi = yo + ky - 2;
      int xi = xbase + m + kx - 2;
      if (yi >= 0 && yi < NN && xi >= 0 && xi < NN) {
        const v8h* src = (const v8h*)(S16 + (((size_t)yi * NN + xi) * BB + b) * OO);
        v8h* dst = (v8h*)(A + m * 160 + kidx * 16);
        dst[0] = src[0]; dst[1] = src[1];
      }
    }
  }
  v8f acc[3]; for (int t = 0; t < 3; ++t) acc[t] = vzero8();
  for (int kk = 0; kk < 5; ++kk) {
    v16h a = ldA16(A + ln * 160, kk * 32, kh);
    for (int t = 0; t < 3; ++t) {
      v16h bm = ldB16(wB1 + (t * 16 + ln) * 160, kk * 32, kh);
      acc[t] = wmma16(a, bm, acc[t]);
    }
  }
  for (int t = 0; t < 3; ++t) {
    int co = t * 16 + ln;
    float bias = cb1[co];
    for (int j = 0; j < 8; ++j) {
      int m = j + kh * 8;
      int xo = xbase + m;
      if (xo < 514) {
        H16[(((size_t)yo * 514 + xo) * BB + b) * CH + co] =
            (_Float16)gelu_exact(acc[t][j] + bias);
      }
    }
  }
}

// ---------------------------------------------------------------------------
// Conv2 (valid) as implicit GEMM: K=9*48 (pad 448), N=16, + bias + |i-j|
// length scaling, writes final S to d_out [512][512][B][16] f32.
// ---------------------------------------------------------------------------
#define C2_WAVES 4
__global__ __launch_bounds__(128) void conv2_kernel(
    const _Float16* __restrict__ H16, const _Float16* __restrict__ wB2,
    const float* __restrict__ cb2, float* __restrict__ outS) {
  __shared__ _Float16 sA[C2_WAVES][16 * 448];
  const int wave = threadIdx.x >> 5;
  const int lane = threadIdx.x & 31;
  const int kh = lane >> 4, ln = lane & 15;
  int wid = blockIdx.x * C2_WAVES + wave;  // exact: 8192*4 = 512*32*2
  int b = wid & 1;
  int rest = wid >> 1;
  int xt = rest & 31;
  int y = rest >> 5;
  int xbase = xt * 16;
  _Float16* A = sA[wave];

  for (int i = 0; i < 8; ++i) {           // zero K tail 432..447
    int idx = i * 32 + lane;
    int m = idx >> 4, k = 432 + (idx & 15);
    A[m * 448 + k] = (_Float16)0.0f;
  }
  for (int i = 0; i < 5; ++i) {           // 144 im2col chunk jobs, 48 ch each
    int j = i * 32 + lane;
    if (j < 144) {
      int m = j / 9, kidx = j % 9;
      int ky = kidx / 3, kx = kidx % 3;
      int yh = y + ky, xh = xbase + m + kx;  // always in [0,514)
      const v8h* src = (const v8h*)(H16 + (((size_t)yh * 514 + xh) * BB + b) * CH);
      v8h* dst = (v8h*)(A + m * 448 + kidx * 48);
      for (int q = 0; q < 6; ++q) dst[q] = src[q];
    }
  }
  v8f acc = vzero8();
  for (int kk = 0; kk < 14; ++kk) {
    v16h a = ldA16(A + ln * 448, kk * 32, kh);
    v16h bm = ldB16(wB2 + ln * 448, kk * 32, kh);
    acc = wmma16(a, bm, acc);
  }
  float bias = cb2[ln];
  for (int j = 0; j < 8; ++j) {
    int m = j + kh * 8;
    int xx = xbase + m;
    int diff = y - xx; if (diff < 0) diff = -diff; if (diff < 1) diff = 1;
    outS[(((size_t)y * NN + xx) * BB + b) * OO + ln] =
        (float)diff * (acc[j] + bias);
  }
}

// ---------------------------------------------------------------------------
// Skip MLP on adjacent frames: [511*B] rows, 384->64->64->16
// ---------------------------------------------------------------------------
#define SK_WAVES 4
__global__ __launch_bounds__(128) void skip_mlp_kernel(
    const float* __restrict__ x,
    const _Float16* __restrict__ sw1h, const float* __restrict__ sb1,
    const _Float16* __restrict__ sw2h, const float* __restrict__ sb2,
    const _Float16* __restrict__ sw3h, const float* __restrict__ sb3,
    float* __restrict__ outSkip) {
  __shared__ _Float16 sA[SK_WAVES][16 * 128];
  __shared__ _Float16 sH[SK_WAVES][16 * 64];
  const int wave = threadIdx.x >> 5;
  const int lane = threadIdx.x & 31;
  const int kh = lane >> 4, ln = lane & 15;
  const int tile = blockIdx.x * SK_WAVES + wave;  // 0..63 (1022 rows, pad)
  _Float16* A = sA[wave];
  _Float16* Hb = sH[wave];

  v8f acc[4]; for (int t = 0; t < 4; ++t) acc[t] = vzero8();
  for (int seg = 0; seg < 3; ++seg) {
    for (int i = 0; i < 64; ++i) {
      int idx = i * 32 + lane;
      int m = idx >> 7, c = idx & 127;
      int gidx = tile * 16 + m;
      int t = gidx >> 1; if (t > 510) t = 510;  // clamp padded rows
      int b = m & 1;
      float xa = x[(t * BB + b) * DD + c];
      float xb = x[((t + 1) * BB + b) * DD + c];
      float v = (seg == 0) ? xa : ((seg == 1) ? xb : xa * xb);
      A[m * 128 + c] = (_Float16)v;
    }
    for (int kk = 0; kk < 4; ++kk) {
      v16h a = ldA16(A + ln * 128, kk * 32, kh);
      for (int t = 0; t < 4; ++t) {
        v16h bm = ldB16(sw1h + (t * 16 + ln) * 384, seg * 128 + kk * 32, kh);
        acc[t] = wmma16(a, bm, acc[t]);
      }
    }
  }
  for (int t = 0; t < 4; ++t) {
    int n = t * 16 + ln;
    float bias = sb1[n];
    for (int j = 0; j < 8; ++j)
      Hb[(j + kh * 8) * 64 + n] = (_Float16)gelu_exact(acc[t][j] + bias);
  }
  v8f acc2[4]; for (int t = 0; t < 4; ++t) acc2[t] = vzero8();
  for (int kk = 0; kk < 2; ++kk) {
    v16h a = ldA16(Hb + ln * 64, kk * 32, kh);
    for (int t = 0; t < 4; ++t) {
      v16h bm = ldB16(sw2h + (t * 16 + ln) * 64, kk * 32, kh);
      acc2[t] = wmma16(a, bm, acc2[t]);
    }
  }
  for (int t = 0; t < 4; ++t) {
    int n = t * 16 + ln;
    float bias = sb2[n];
    for (int j = 0; j < 8; ++j)
      Hb[(j + kh * 8) * 64 + n] = (_Float16)gelu_exact(acc2[t][j] + bias);
  }
  v8f acc3 = vzero8();
  for (int kk = 0; kk < 2; ++kk) {
    v16h a = ldA16(Hb + ln * 64, kk * 32, kh);
    v16h bm = ldB16(sw3h + ln * 64, kk * 32, kh);
    acc3 = wmma16(a, bm, acc3);
  }
  float bias3 = sb3[ln];
  for (int j = 0; j < 8; ++j) {
    int m = j + kh * 8;
    int gidx = tile * 16 + m;
    if (gidx < (NN - 1) * BB)
      outSkip[gidx * OO + ln] = acc3[j] + bias3;
  }
}

// ---------------------------------------------------------------------------
// Launch
// ---------------------------------------------------------------------------
extern "C" void kernel_launch(void* const* d_in, const int* in_sizes, int n_in,
                              void* d_out, int out_size, void* d_ws,
                              size_t ws_size, hipStream_t stream) {
  (void)in_sizes; (void)n_in; (void)out_size; (void)ws_size;
  const float* x   = (const float*)d_in[0];
  const float* w1  = (const float*)d_in[1];
  const float* b1  = (const float*)d_in[2];
  const float* w2  = (const float*)d_in[3];
  const float* b2  = (const float*)d_in[4];
  const float* w3  = (const float*)d_in[5];
  const float* b3  = (const float*)d_in[6];
  const float* sw1 = (const float*)d_in[7];
  const float* sb1 = (const float*)d_in[8];
  const float* sw2 = (const float*)d_in[9];
  const float* sb2 = (const float*)d_in[10];
  const float* sw3 = (const float*)d_in[11];
  const float* sb3 = (const float*)d_in[12];
  const float* cw1 = (const float*)d_in[13];
  const float* cb1 = (const float*)d_in[14];
  const float* cw2 = (const float*)d_in[15];
  const float* cb2 = (const float*)d_in[16];

  float* outS = (float*)d_out;
  float* outSkip = outS + (size_t)NN * NN * BB * OO;

  char* ws = (char*)d_ws;
  size_t off = 0;
  auto walloc = [&](size_t bytes) -> void* {
    void* p = ws + off;
    off += (bytes + 255) & ~(size_t)255;
    return p;
  };
  float*    xc   = (float*)walloc((size_t)513 * BB * DD * 4);
  float*    x2c  = (float*)walloc((size_t)513 * BB * DD * 4);
  float*    x3c  = (float*)walloc((size_t)513 * BB * DD * 4);
  _Float16* w1h  = (_Float16*)walloc((size_t)HH * 6 * DD * 2);
  _Float16* w2h  = (_Float16*)walloc((size_t)HH * HH * 2);
  _Float16* w3h  = (_Float16*)walloc((size_t)OO * HH * 2);
  _Float16* sw1h = (_Float16*)walloc((size_t)HH * 3 * DD * 2);
  _Float16* sw2h = (_Float16*)walloc((size_t)HH * HH * 2);
  _Float16* sw3h = (_Float16*)walloc((size_t)OO * HH * 2);
  _Float16* wB1  = (_Float16*)walloc((size_t)CH * 160 * 2);
  _Float16* wB2  = (_Float16*)walloc((size_t)OO * 448 * 2);
  _Float16* S16  = (_Float16*)walloc((size_t)NN * NN * BB * OO * 2);
  _Float16* H16  = (_Float16*)walloc((size_t)514 * 514 * BB * CH * 2);

  prep_weights_kernel<<<192, 256, 0, stream>>>(
      w1, w2, w3, sw1, sw2, sw3, cw1, cw2,
      w1h, w2h, w3h, sw1h, sw2h, sw3h, wB1, wB2);
  prefix_kernel<<<1, 256, 0, stream>>>(x, xc, x2c, x3c);
  zero32_kernel<<<4096, 256, 0, stream>>>(
      (unsigned int*)S16, (unsigned int)((size_t)NN * NN * BB * OO / 2));
  pair_mlp_kernel<<<MROWS / 16 / PM_WAVES, 256, 0, stream>>>(
      x, xc, x2c, x3c, w1h, b1, w2h, b2, w3h, b3, S16);
  conv1_kernel<<<514 * 33 * BB / C1_WAVES, 128, 0, stream>>>(S16, wB1, cb1, H16);
  conv2_kernel<<<NN * 32 * BB / C2_WAVES, 128, 0, stream>>>(H16, wB2, cb2, outS);
  skip_mlp_kernel<<<64 / SK_WAVES, 128, 0, stream>>>(
      x, sw1h, sb1, sw2h, sb2, sw3h, sb3, outSkip);
}
